// GCN_64957085384720
// MI455X (gfx1250) — matrix operand order repaired and checked by reading, not compiled, not run on hardware
//
#include <hip/hip_runtime.h>
#include <math.h>

typedef __attribute__((ext_vector_type(16))) __bf16 v16bf;
typedef __attribute__((ext_vector_type(8)))  __bf16 v8bf;
typedef __attribute__((ext_vector_type(8)))  float  v8f;
typedef __attribute__((ext_vector_type(4)))  int    v4i;

#define HID 2048
#define IN_C 300
#define KPAD 320
#define NEG_SLOPE 0.2f

#define AS1 __attribute__((address_space(1)))
#define AS3 __attribute__((address_space(3)))

#if __has_builtin(__builtin_amdgcn_global_load_async_to_lds_b128)
#define USE_ASYNC_LDS 1
#else
#define USE_ASYNC_LDS 0
#endif

// ------------------------------------------------------------------
// Degree / normalization helpers
// ------------------------------------------------------------------
__global__ void deg_kernel(const int* __restrict__ ei, float* __restrict__ deg, int E) {
  int e = blockIdx.x * blockDim.x + threadIdx.x;
  if (e < E) atomicAdd(&deg[ei[2 * e + 1]], 1.0f);
}

__global__ void invdeg_kernel(float* deg, int N) {
  int i = blockIdx.x * blockDim.x + threadIdx.x;
  if (i < N) deg[i] = 1.0f / fmaxf(deg[i], 1.0f);
}

// ------------------------------------------------------------------
// fp32 -> bf16 conversions (weights transposed to N x K for GEMM B)
// ------------------------------------------------------------------
__global__ void conv_x_kernel(const float* __restrict__ x, __bf16* __restrict__ xb, int N) {
  int idx = blockIdx.x * blockDim.x + threadIdx.x;
  if (idx >= N * KPAD) return;
  int n = idx / KPAD, k = idx - n * KPAD;
  float v = (k < IN_C) ? x[(size_t)n * IN_C + k] : 0.0f;
  xb[idx] = (__bf16)v;
}

__global__ void conv_wT_kernel(const float* __restrict__ w, __bf16* __restrict__ wt,
                               int Ksrc, int Kpad, int Nmat) {
  int idx = blockIdx.x * blockDim.x + threadIdx.x;
  if (idx >= Nmat * Kpad) return;
  int n = idx / Kpad, k = idx - n * Kpad;
  float v = (k < Ksrc) ? w[(size_t)k * Nmat + n] : 0.0f;
  wt[idx] = (__bf16)v;
}

// ------------------------------------------------------------------
// bias + leaky ReLU + downcast hidden state to bf16 (GEMM2 A operand)
// ------------------------------------------------------------------
__global__ void bias_lrelu_bf16_kernel(const float* __restrict__ h, const float* __restrict__ b,
                                       __bf16* __restrict__ out, long long total) {
  long long idx = (long long)blockIdx.x * blockDim.x + threadIdx.x;
  if (idx >= total) return;
  int c = (int)(idx & (HID - 1));
  float v = h[idx] + b[c];
  v = (v > 0.0f) ? v : NEG_SLOPE * v;
  out[idx] = (__bf16)v;
}

// ------------------------------------------------------------------
// SpMM: out[row] += invdeg[row] * in[col]   (edge-parallel, f32 atomics)
// grid = (E, HID/1024), block = 256, 4 floats/thread
// ------------------------------------------------------------------
__global__ __launch_bounds__(256) void spmm_kernel(const int* __restrict__ ei,
                                                   const float* __restrict__ invdeg,
                                                   const float* __restrict__ in,
                                                   float* __restrict__ out) {
  int e = blockIdx.x;
  int f = (blockIdx.y * blockDim.x + threadIdx.x) * 4;
  int col = ei[2 * e + 0];
  int row = ei[2 * e + 1];
  float v = invdeg[row];
  const float4 hv = *(const float4*)(in + (size_t)col * HID + f);
  float* o = out + (size_t)row * HID + f;
  atomicAdd(o + 0, v * hv.x);
  atomicAdd(o + 1, v * hv.y);
  atomicAdd(o + 2, v * hv.z);
  atomicAdd(o + 3, v * hv.w);
}

// ------------------------------------------------------------------
// async global -> LDS copy helpers (16 bytes per call)
// ------------------------------------------------------------------
__device__ __forceinline__ void copy16(__bf16* lds, const __bf16* g) {
#if USE_ASYNC_LDS
  __builtin_amdgcn_global_load_async_to_lds_b128(
      (AS1 v4i*)(AS1 void*)(g), (AS3 v4i*)(AS3 void*)(lds), 0, 0);
#else
  *(v8bf*)lds = *(const v8bf*)g;
#endif
}

__device__ __forceinline__ void wait_async_copies() {
#if USE_ASYNC_LDS
#if __has_builtin(__builtin_amdgcn_s_wait_asynccnt)
  __builtin_amdgcn_s_wait_asynccnt(0);
#else
  asm volatile("s_wait_asynccnt 0x0" ::: "memory");
#endif
#endif
}

// ------------------------------------------------------------------
// bf16 WMMA GEMM:  C[M x N] = A[M x K] * B^T[N x K]
// Block = 256 threads (8 waves). Block tile 128x128, wave tile 32x64
// (2x4 fragments of v_wmma_f32_16x16x32_bf16). K multiple of 32.
// Double-buffered LDS with async global->LDS staging of tile k+1
// overlapped with WMMA on tile k.
// ------------------------------------------------------------------
#define TM 128
#define TN 128
#define TK 32
#define LDSS 40  // bf16 elements per LDS row; padded to break bank conflicts

__global__ __launch_bounds__(256) void gemm_bf16_wmma(const __bf16* __restrict__ A,
                                                      const __bf16* __restrict__ BT,
                                                      float* __restrict__ C,
                                                      int M, int N, int K) {
  __shared__ __bf16 As[2][TM * LDSS];
  __shared__ __bf16 Bs[2][TN * LDSS];

  const int tid  = threadIdx.x;
  const int lane = tid & 31;
  const int wave = tid >> 5;
  const int bm = blockIdx.y * TM;
  const int bn = blockIdx.x * TN;
  const int wm = (wave & 3) * 32;   // wave row offset within block tile
  const int wn = (wave >> 2) * 64;  // wave col offset within block tile

  // staging coordinates: each thread copies 2x16B of A and 2x16B of B
  const int sr  = tid >> 1;          // 0..127 tile row
  const int skk = (tid & 1) * 16;    // 0 / 16 k offset
  int a_row = bm + sr;
  if (a_row >= M) a_row = M - 1;     // clamp: branch-free staging, stores guarded
  const __bf16* aptr = A + (size_t)a_row * K + skk;
  const __bf16* bptr = BT + (size_t)(bn + sr) * K + skk;
  __bf16* adst0 = &As[0][sr * LDSS + skk];
  __bf16* bdst0 = &Bs[0][sr * LDSS + skk];
  const int bufstride_a = TM * LDSS;
  const int bufstride_b = TN * LDSS;

  auto stage = [&](int buf, int k0) {
    __bf16* ad = adst0 + buf * bufstride_a;
    __bf16* bd = bdst0 + buf * bufstride_b;
    copy16(ad,     aptr + k0);
    copy16(ad + 8, aptr + k0 + 8);
    copy16(bd,     bptr + k0);
    copy16(bd + 8, bptr + k0 + 8);
  };

  v8f acc[2][4] = {};

  stage(0, 0);
  wait_async_copies();
  __syncthreads();

  // fragment coordinates per documented CDNA5 VGPR layouts
  // A 16x32: lanes 0-15 hold M=lane, K {0..7,16..23}; lanes 16-31 K {8..15,24..31}
  const int am_base = wm + (lane & 15);
  const int kbA = (lane < 16) ? 0 : 8;
  // B 32x16: lanes 0-15 hold N=lane, K {0..7,16..23}; lanes 16-31 K {8..15,24..31}
  const int bn_base = wn + (lane & 15);
  const int kbB = (lane < 16) ? 0 : 8;

  int buf = 0;
  for (int k0 = 0; k0 < K; k0 += TK) {
    // kick off async staging of the next tile into the other buffer
    if (k0 + TK < K) stage(buf ^ 1, k0 + TK);
    if (k0 + 2 * TK < K) {
      __builtin_prefetch(aptr + k0 + 2 * TK, 0, 1);
      __builtin_prefetch(bptr + k0 + 2 * TK, 0, 1);
    }

    v16bf afrag[2], bfrag[4];
#pragma unroll
    for (int i = 0; i < 2; ++i) {
      const __bf16* ap = &As[buf][(am_base + i * 16) * LDSS];
      *(((v8bf*)&afrag[i]) + 0) = *(const v8bf*)(ap + kbA);
      *(((v8bf*)&afrag[i]) + 1) = *(const v8bf*)(ap + kbA + 16);
    }
#pragma unroll
    for (int j = 0; j < 4; ++j) {
      const __bf16* bp = &Bs[buf][(bn_base + j * 16) * LDSS + kbB];
      *(((v8bf*)&bfrag[j]) + 0) = *(const v8bf*)(bp);
      *(((v8bf*)&bfrag[j]) + 1) = *(const v8bf*)(bp + 16);
    }

#pragma unroll
    for (int i = 0; i < 2; ++i)
#pragma unroll
      for (int j = 0; j < 4; ++j)
        acc[i][j] = __builtin_amdgcn_wmma_f32_16x16x32_bf16(
            false, afrag[i], false, bfrag[j], (short)0, acc[i][j], false, false);

    wait_async_copies();  // next buffer fully landed in LDS
    __syncthreads();      // everyone done reading current buffer
    buf ^= 1;
  }

  // epilogue: C/D layout: VGPR j -> M=j (lanes 0-15) / M=j+8 (lanes 16-31), N=lane&15
  const int mhalf = (lane >= 16) ? 8 : 0;
#pragma unroll
  for (int i = 0; i < 2; ++i) {
    const int rbase = bm + wm + i * 16 + mhalf;
#pragma unroll
    for (int j = 0; j < 4; ++j) {
      const int col = bn + wn + j * 16 + (lane & 15);
#pragma unroll
      for (int e = 0; e < 8; ++e) {
        const int row = rbase + e;
        if (row < M) C[(size_t)row * N + col] = acc[i][j][e];
      }
    }
  }
}

// ------------------------------------------------------------------
// bias + row L2 normalize: one block per row (2048 cols, 8/thread)
// ------------------------------------------------------------------
__global__ __launch_bounds__(256) void norm_kernel(const float* __restrict__ h,
                                                   const float* __restrict__ b2,
                                                   float* __restrict__ out) {
  const int row = blockIdx.x;
  const float* hr = h + (size_t)row * HID;
  float vals[8];
  float ss = 0.f;
#pragma unroll
  for (int i = 0; i < 8; ++i) {
    int c = i * 256 + threadIdx.x;
    float t = hr[c] + b2[c];
    vals[i] = t;
    ss += t * t;
  }
#pragma unroll
  for (int off = 16; off > 0; off >>= 1) ss += __shfl_xor(ss, off, 32);
  __shared__ float red[8];
  if ((threadIdx.x & 31) == 0) red[threadIdx.x >> 5] = ss;
  __syncthreads();
  float tot = 0.f;
#pragma unroll
  for (int w = 0; w < 8; ++w) tot += red[w];
  const float inv = 1.0f / fmaxf(sqrtf(tot), 1e-12f);
  float* orow = out + (size_t)row * HID;
#pragma unroll
  for (int i = 0; i < 8; ++i) orow[i * 256 + threadIdx.x] = vals[i] * inv;
}

// ------------------------------------------------------------------
// Host orchestration
// ------------------------------------------------------------------
extern "C" void kernel_launch(void* const* d_in, const int* in_sizes, int n_in,
                              void* d_out, int out_size, void* d_ws, size_t ws_size,
                              hipStream_t stream) {
  const float* x  = (const float*)d_in[0];
  const int*   ei = (const int*)d_in[1];
  const float* w1 = (const float*)d_in[2];
  const float* b1 = (const float*)d_in[3];
  const float* w2 = (const float*)d_in[4];
  const float* b2 = (const float*)d_in[5];

  const int N = in_sizes[0] / IN_C;
  const int E = in_sizes[1] / 2;

  const size_t hbytes = (size_t)N * HID * sizeof(float);

  // workspace layout (256B aligned chunks)
  char* ws = (char*)d_ws;
  size_t off = 0;
  auto take = [&](size_t bytes) -> void* {
    void* p = ws + off;
    off = (off + bytes + 255) & ~(size_t)255;
    return p;
  };
  float*  deg  = (float*)take((size_t)N * sizeof(float));
  float*  bufA = (float*)take(hbytes);
  float*  bufB = (float*)take(hbytes);
  __bf16* Hb   = (__bf16*)take((size_t)N * HID * sizeof(__bf16));
  __bf16* W2T  = (__bf16*)take((size_t)HID * HID * sizeof(__bf16));
  // one-shot bf16 conversions of x / w1 alias into bufB (free until SpMM #1)
  __bf16* Xb  = (__bf16*)bufB;
  __bf16* W1T = (__bf16*)((char*)bufB + (((size_t)N * KPAD * sizeof(__bf16) + 255) & ~(size_t)255));

  const int B256 = 256;

  // 1) degrees -> inverse degrees
  (void)hipMemsetAsync(deg, 0, (size_t)N * sizeof(float), stream);
  deg_kernel<<<(E + B256 - 1) / B256, B256, 0, stream>>>(ei, deg, E);
  invdeg_kernel<<<(N + B256 - 1) / B256, B256, 0, stream>>>(deg, N);

  // 2) bf16 conversions (weights transposed to N x K)
  conv_x_kernel<<<((size_t)N * KPAD + B256 - 1) / B256, B256, 0, stream>>>(x, Xb, N);
  conv_wT_kernel<<<((size_t)HID * KPAD + B256 - 1) / B256, B256, 0, stream>>>(w1, W1T, IN_C, KPAD, HID);
  conv_wT_kernel<<<((size_t)HID * HID + B256 - 1) / B256, B256, 0, stream>>>(w2, W2T, HID, HID, HID);

  const dim3 gemm_grid(HID / TN, (N + TM - 1) / TM);

  // 3) h = x @ w1   (bf16 WMMA, f32 accumulate)
  gemm_bf16_wmma<<<gemm_grid, B256, 0, stream>>>(Xb, W1T, bufA, N, HID, KPAD);

  const dim3 spmm_grid((unsigned)E, HID / (B256 * 4));
  // 4) two propagation steps
  (void)hipMemsetAsync(bufB, 0, hbytes, stream);
  spmm_kernel<<<spmm_grid, B256, 0, stream>>>(ei, deg, bufA, bufB);
  (void)hipMemsetAsync(bufA, 0, hbytes, stream);
  spmm_kernel<<<spmm_grid, B256, 0, stream>>>(ei, deg, bufB, bufA);

  // 5) bias + leaky relu + bf16
  bias_lrelu_bf16_kernel<<<((size_t)N * HID + B256 - 1) / B256, B256, 0, stream>>>(
      bufA, b1, Hb, (long long)N * HID);

  // 6) h = h @ w2
  gemm_bf16_wmma<<<gemm_grid, B256, 0, stream>>>(Hb, W2T, bufA, N, HID, HID);

  // 7) two propagation steps
  (void)hipMemsetAsync(bufB, 0, hbytes, stream);
  spmm_kernel<<<spmm_grid, B256, 0, stream>>>(ei, deg, bufA, bufB);
  (void)hipMemsetAsync(bufA, 0, hbytes, stream);
  spmm_kernel<<<spmm_grid, B256, 0, stream>>>(ei, deg, bufB, bufA);

  // 8) bias + row L2 normalize -> output
  norm_kernel<<<N, B256, 0, stream>>>(bufA, b2, (float*)d_out);
}
